// dynamic_model_9139690406284
// MI455X (gfx1250) — matrix operand order repaired and checked by reading, not compiled, not run on hardware
//
#include <hip/hip_runtime.h>
#include <hip/hip_bf16.h>

typedef __attribute__((ext_vector_type(16))) _Float16 v16h;
typedef __attribute__((ext_vector_type(8)))  float    v8f;

__device__ __forceinline__ float sigmoidf_(float x){ return 1.0f/(1.0f+expf(-x)); }

__device__ __forceinline__ v8f wmma_f32_16x16x32_f16(v16h a, v16h b, v8f c){
  // 8 args: (neg_a, A, neg_b, B, c_mod, C, reuse_a, reuse_b)
  return __builtin_amdgcn_wmma_f32_16x16x32_f16(false, a, false, b, (short)0, c, false, false);
}

#define ADJS 264   // padded adjacency row stride (halves): 264*2B = 528B, staggers banks
#define PHS  264   // padded phiT row stride (halves)

// =====================================================================
// Kernel 1: process_frame for the 10 encoder frames (gt frames are dead
// code in the reference decoder). One workgroup per (batch, frame).
// Full 256x256 f16 adjacency kept in LDS (~132KB of 320KB WGP LDS).
// GEMMs via v_wmma_f32_16x16x32_f16 with f32 accumulation.
// =====================================================================
__global__ __launch_bounds__(256, 1) void frame_gcn_kernel(
    const float* __restrict__ X,
    const float* __restrict__ thW, const float* __restrict__ thb,
    const float* __restrict__ phW, const float* __restrict__ phb,
    const float* __restrict__ g1W, const float* __restrict__ g1b,
    const float* __restrict__ g2W, const float* __restrict__ g2b,
    const float* __restrict__ ln_g, const float* __restrict__ ln_b,
    const float* __restrict__ pW,  const float* __restrict__ pb,
    float* __restrict__ feats)
{
  const int b    = blockIdx.x;   // batch 0..31
  const int f    = blockIdx.y;   // frame 0..9
  const int p    = f;            // per-frame param index
  const int tid  = threadIdx.x;  // 0..255
  const int wave = tid >> 5;     // 8 waves (wave32)
  const int lane = tid & 31;

  extern __shared__ char smem[];
  float*    s_frame = (float*)smem;               // [256][10]
  float*    s_pos   = s_frame + 2560;             // [256][2]
  float*    s_r     = s_pos   + 512;              // [256]
  float*    s_dinv  = s_r     + 256;              // [256]
  float*    s_x1    = s_dinv  + 256;              // [256][16] (Hraw, then x1)
  float*    s_x2    = s_x1    + 4096;             // [256][8]
  float*    s_red   = s_x2    + 2048;             // [256]
  float*    s_red2  = s_red   + 256;              // [256]
  float*    s_sc    = s_red2  + 256;              // [4]
  _Float16* s_theta = (_Float16*)(s_sc + 4);      // [256][32] (K padded)
  _Float16* s_phiT  = s_theta + 256*32;           // [32][PHS]: phiT[k][n]
  _Float16* s_h1    = s_phiT  + 32*PHS;           // [256][16]
  _Float16* s_yv    = s_h1    + 256*16;           // [256][16]
  _Float16* s_adj   = s_yv    + 256*16;           // [256][ADJS]

  // ---- Stage 1: load node row; theta (f16, K padded), phiT (transposed), Hraw ----
  {
    const float* fr = X + (((size_t)b * 20 + f) * 256 + tid) * 10;
    float x[10];
    #pragma unroll
    for (int d = 0; d < 10; ++d) { x[d] = fr[d]; s_frame[tid*10 + d] = x[d]; }
    float px = x[1] * 1000.f, py = x[2] * 1000.f;
    s_pos[tid*2] = px; s_pos[tid*2+1] = py; s_r[tid] = px*px + py*py;
    #pragma unroll
    for (int j = 0; j < 16; ++j) {
      float ta = thb[p*16 + j], ph = phb[p*16 + j], hh = 0.f;
      #pragma unroll
      for (int d = 0; d < 10; ++d) {
        ta += x[d] * thW[(p*16 + j)*10 + d];
        ph += x[d] * phW[(p*16 + j)*10 + d];
        hh += x[d] * g1W[(p*10 + d)*16 + j];   // frame @ g1W
      }
      s_theta[tid*32 + j]      = (_Float16)ta;
      s_theta[tid*32 + 16 + j] = (_Float16)0.f;
      s_phiT[j*PHS + tid]        = (_Float16)ph;   // transposed: [k][n]
      s_phiT[(16 + j)*PHS + tid] = (_Float16)0.f;  // K pad rows
      s_x1[tid*16 + j] = hh;                   // Hraw
    }
  }
  __syncthreads();

  // ---- Stage 2: rel = theta @ phi^T / 4, distance mask -> s_adj (f16) ----
  // 256 tiles of 16x16; each wave owns 32 tiles; one WMMA per tile (K=32 pad).
  for (int tile = wave; tile < 256; tile += 8) {
    const int tm = tile >> 4, tn = tile & 15;
    const int arow = tm*16 + (lane & 15);
    const int kb   = (lane >> 4) << 3;      // A layout: lanes 0-15 K {0-7,16-23}; 16-31 K {8-15,24-31}
    v16h a, bv;
    #pragma unroll
    for (int j = 0; j < 8; ++j) {
      int k = kb + ((j & 3) << 1) + ((j >> 2) << 4);
      a[2*j]   = s_theta[arow*32 + k];
      a[2*j+1] = s_theta[arow*32 + k + 1];
    }
    #pragma unroll
    for (int e = 0; e < 16; ++e)            // B layout: lane = N, elements = K (same order as A)
      bv[e] = s_phiT[(kb + (e & 7) + ((e >> 3) << 4))*PHS + tn*16 + (lane & 15)];
    v8f c = {};
    c = wmma_f32_16x16x32_f16(a, bv, c);
    const int m = tn*16 + (lane & 15);      // D: N on lanes
    #pragma unroll
    for (int j = 0; j < 8; ++j) {
      int n = tm*16 + ((lane >> 4) << 3) + j;  // D: M = j + 8*(lane>=16)
      float v = c[j] * 0.25f;
      float d2 = s_r[n] + s_r[m]
               - 2.f * (s_pos[n*2]*s_pos[m*2] + s_pos[n*2+1]*s_pos[m*2+1]);
      // sqrt(d2) > 30  <=>  d2 > 900 (identical, incl. NaN for d2<0 -> unmasked)
      if (d2 > 900.f) v = -1.0e4f;             // masked logit (exp underflows to 0)
      s_adj[n*ADJS + m] = (_Float16)v;
    }
  }
  __syncthreads();

  // ---- Stage 3: row softmax (relu is a no-op on softmax), diag=1, dinv ----
  {
    float mx = -3.0e38f;
    for (int m = 0; m < 256; ++m) mx = fmaxf(mx, (float)s_adj[tid*ADJS + m]);
    float sum = 0.f;
    for (int m = 0; m < 256; ++m) {            // store exponentials in place
      float e = expf((float)s_adj[tid*ADJS + m] - mx);
      s_adj[tid*ADJS + m] = (_Float16)e;
      sum += e;
    }
    float inv = 1.f / sum, rsum = 0.f;
    for (int m = 0; m < 256; ++m) {
      float vv = (m == tid) ? 1.f : (float)s_adj[tid*ADJS + m] * inv;
      s_adj[tid*ADJS + m] = (_Float16)vv;
      rsum += vv;
    }
    float di = rsqrtf(fmaxf(rsum, 1.f));
    s_dinv[tid] = di;
    // Fold dinv[m] into B rows of GCN1: h1[m] = dinv[m] * Hraw[m]
    #pragma unroll
    for (int j = 0; j < 16; ++j)
      s_h1[tid*16 + j] = (_Float16)(s_x1[tid*16 + j] * di);
  }
  __syncthreads();

  // ---- Stage 4: GCN1  x1 = relu(dinv[n]*(adj @ h1) + g1b)  (M=256,N=16,K=256) ----
  for (int tm = wave; tm < 16; tm += 8) {
    v8f c = {};
    const int arow = tm*16 + (lane & 15);
    const int kb   = (lane >> 4) << 3;
    for (int kk = 0; kk < 8; ++kk) {
      v16h a, bv;
      #pragma unroll
      for (int j = 0; j < 8; ++j) {
        int k = kk*32 + kb + ((j & 3) << 1) + ((j >> 2) << 4);
        a[2*j]   = s_adj[arow*ADJS + k];
        a[2*j+1] = s_adj[arow*ADJS + k + 1];
      }
      #pragma unroll
      for (int e = 0; e < 16; ++e)
        bv[e] = s_h1[(kk*32 + kb + (e & 7) + ((e >> 3) << 4))*16 + (lane & 15)];
      c = wmma_f32_16x16x32_f16(a, bv, c);
    }
    const int col = lane & 15;
    #pragma unroll
    for (int j = 0; j < 8; ++j) {
      int n = tm*16 + ((lane >> 4) << 3) + j;
      float vv = s_dinv[n] * c[j] + g1b[p*16 + col];
      s_x1[n*16 + col] = fmaxf(vv, 0.f);
    }
  }
  __syncthreads();

  // ---- Stage 5: y = (x1 @ g2W) * dinv[m] -> f16, N padded 8->16 ----
  #pragma unroll
  for (int j = 0; j < 8; ++j) {
    float acc = 0.f;
    #pragma unroll
    for (int k = 0; k < 16; ++k) acc += s_x1[tid*16 + k] * g2W[(p*16 + k)*8 + j];
    s_yv[tid*16 + j]     = (_Float16)(acc * s_dinv[tid]);
    s_yv[tid*16 + 8 + j] = (_Float16)0.f;
  }
  __syncthreads();

  // ---- Stage 6: GCN2  x2 = dinv[n]*(adj @ y) + g2b  (N=8 padded to 16) ----
  for (int tm = wave; tm < 16; tm += 8) {
    v8f c = {};
    const int arow = tm*16 + (lane & 15);
    const int kb   = (lane >> 4) << 3;
    for (int kk = 0; kk < 8; ++kk) {
      v16h a, bv;
      #pragma unroll
      for (int j = 0; j < 8; ++j) {
        int k = kk*32 + kb + ((j & 3) << 1) + ((j >> 2) << 4);
        a[2*j]   = s_adj[arow*ADJS + k];
        a[2*j+1] = s_adj[arow*ADJS + k + 1];
      }
      #pragma unroll
      for (int e = 0; e < 16; ++e)
        bv[e] = s_yv[(kk*32 + kb + (e & 7) + ((e >> 3) << 4))*16 + (lane & 15)];
      c = wmma_f32_16x16x32_f16(a, bv, c);
    }
    const int col = lane & 15;
    if (col < 8) {
      #pragma unroll
      for (int j = 0; j < 8; ++j) {
        int n = tm*16 + ((lane >> 4) << 3) + j;
        s_x2[n*8 + col] = s_dinv[n] * c[j] + g2b[p*8 + col];
      }
    }
  }
  __syncthreads();

  // ---- Stage 7: LayerNorm stats over all [256,8] ----
  {
    float sm = 0.f, sq = 0.f;
    #pragma unroll
    for (int j = 0; j < 8; ++j) { float vv = s_x2[tid*8 + j]; sm += vv; sq += vv*vv; }
    s_red[tid] = sm; s_red2[tid] = sq;
  }
  __syncthreads();
  if (tid == 0) {
    float S = 0.f, Q = 0.f;
    for (int i = 0; i < 256; ++i) { S += s_red[i]; Q += s_red2[i]; }
    float mu  = S / 2048.f;
    float var = Q / 2048.f - mu*mu;
    s_sc[0] = mu; s_sc[1] = rsqrtf(var + 1e-5f);
  }
  __syncthreads();

  // ---- Stage 8: node-0 feature + agent head ----
  if (tid < 8) {
    float xv = (s_x2[tid] - s_sc[0]) * s_sc[1] * ln_g[p*2048 + tid] + ln_b[p*2048 + tid];
    xv = fmaxf(xv, 0.f);
    float ag = pb[p*8 + tid];
    #pragma unroll
    for (int d = 0; d < 10; ++d) ag += s_frame[d] * pW[(p*8 + tid)*10 + d];
    ag = fmaxf(ag, 0.f);
    feats[((size_t)f*32 + b)*8 + tid] = xv * 0.1f + ag;
  }
}

// =====================================================================
// Kernel 2: bidirectional 2-layer LSTM encoder + attention decoder.
// Tiny & sequential: one workgroup, 256 threads = (batch 32) x (hidden 8).
// =====================================================================
struct SeqParams {
  const float *e0fWhh,*e0fWih,*e0fbhh,*e0fbih;
  const float *e0bWhh,*e0bWih,*e0bbhh,*e0bbih;
  const float *e1fWhh,*e1fWih,*e1fbhh,*e1fbih;
  const float *e1bWhh,*e1bWih,*e1bbhh,*e1bbih;
  const float *attn_W,*attn_b,*vv,*fc_W,*fc_b;
  const float *d0Whh,*d0Wih,*d0bhh,*d0bih;
  const float *d1Whh,*d1Wih,*d1bhh,*d1bih;
  const float *fcout_W,*fcout_b;
};

__device__ void lstm_scan_lds(const float* Wih, const float* Whh,
                              const float* bih, const float* bhh,
                              int in_dim, const float* xs, float* ys,
                              float* hfin, float* cfin,
                              float* hbuf, float* cbuf, int reverse)
{
  const int tid = threadIdx.x, bb = tid >> 3, hh = tid & 7;
  hbuf[tid] = 0.f; cbuf[tid] = 0.f;
  __syncthreads();
  for (int s = 0; s < 10; ++s) {
    int t = reverse ? (9 - s) : s;
    float g[4];
    for (int q = 0; q < 4; ++q) {
      int row = q*8 + hh;
      float acc = bih[row] + bhh[row];
      const float* xrow = xs + (t*32 + bb) * in_dim;
      for (int d = 0; d < in_dim; ++d) acc += xrow[d] * Wih[row*in_dim + d];
      for (int j = 0; j < 8; ++j)      acc += hbuf[bb*8 + j] * Whh[row*8 + j];
      g[q] = acc;
    }
    float cprev = cbuf[tid];
    float c2 = sigmoidf_(g[1]) * cprev + sigmoidf_(g[0]) * tanhf(g[2]);
    float h2 = sigmoidf_(g[3]) * tanhf(c2);
    __syncthreads();
    hbuf[tid] = h2; cbuf[tid] = c2; ys[(t*32 + bb)*8 + hh] = h2;
    __syncthreads();
  }
  hfin[tid] = hbuf[tid]; cfin[tid] = cbuf[tid];
  __syncthreads();
}

__global__ __launch_bounds__(256, 1) void seq_kernel(
    const float* __restrict__ feats, SeqParams P, float* __restrict__ out)
{
  const int tid = threadIdx.x, bb = tid >> 3, hh = tid & 7;
  extern __shared__ char smem[];
  float* ef   = (float*)smem;   // [10][32][8]
  float* f0   = ef   + 2560;
  float* b0   = f0   + 2560;
  float* x1c  = b0   + 2560;    // [10][32][16]
  float* f1   = x1c  + 5120;
  float* b1   = f1   + 2560;
  float* eo   = b1   + 2560;    // enc_out [10][32][8]
  float* hid  = eo   + 2560;    // [2][32][8]
  float* cel  = hid  + 512;
  float* hbuf = cel  + 512;
  float* cbuf = hbuf + 256;
  float* hf0  = cbuf + 256; float* cf0 = hf0 + 256;
  float* hb0v = cf0  + 256; float* cb0 = hb0v + 256;
  float* hf1  = cb0  + 256; float* cf1 = hf1 + 256;
  float* hb1v = cf1  + 256; float* cb1 = hb1v + 256;
  float* inp  = cb1  + 256;
  float* hm   = inp  + 256;
  float* xc   = hm   + 256;     // [32][16]
  float* ctx  = xc   + 512;
  float* th0  = ctx  + 256;
  float* pred = th0  + 256;

  for (int i = tid; i < 2560; i += 256) ef[i] = feats[i];
  __syncthreads();

  lstm_scan_lds(P.e0fWih, P.e0fWhh, P.e0fbih, P.e0fbhh, 8,  ef,  f0, hf0,  cf0, hbuf, cbuf, 0);
  lstm_scan_lds(P.e0bWih, P.e0bWhh, P.e0bbih, P.e0bbhh, 8,  ef,  b0, hb0v, cb0, hbuf, cbuf, 1);
  for (int i = tid; i < 5120; i += 256) {
    int t = i / 512, r = i % 512, bq = r / 16, cq = r % 16;
    x1c[i] = (cq < 8) ? f0[(t*32 + bq)*8 + cq] : b0[(t*32 + bq)*8 + cq - 8];
  }
  __syncthreads();
  lstm_scan_lds(P.e1fWih, P.e1fWhh, P.e1fbih, P.e1fbhh, 16, x1c, f1, hf1,  cf1, hbuf, cbuf, 0);
  lstm_scan_lds(P.e1bWih, P.e1bWhh, P.e1bbih, P.e1bbhh, 16, x1c, b1, hb1v, cb1, hbuf, cbuf, 1);

  // enc_out = cat(f1,b1).view(10,2,-1,8).sum(1)  (faithful flat-index trick)
  for (int i = tid; i < 2560; i += 256) {
    int t = i / 256, k = i % 256;
    int b1q = k / 16,        c1q = k % 16;
    int b2q = (k + 256)/16,  c2q = (k + 256) % 16;
    float v1 = (c1q < 8) ? f1[(t*32 + b1q)*8 + c1q] : b1[(t*32 + b1q)*8 + c1q - 8];
    float v2 = (c2q < 8) ? f1[(t*32 + b2q)*8 + c2q] : b1[(t*32 + b2q)*8 + c2q - 8];
    eo[i] = v1 + v2;
  }
  hid[tid] = hf0[tid] + hb0v[tid]; hid[256 + tid] = hf1[tid] + hb1v[tid];
  cel[tid] = cf0[tid] + cb0[tid];  cel[256 + tid] = cf1[tid] + cb1[tid];
  inp[tid] = ef[9*256 + tid];   // trg[0] = enc_feats[-1]
  __syncthreads();

  for (int step = 0; step < 10; ++step) {
    hm[tid] = 0.5f * (hid[tid] + hid[256 + tid]);
    __syncthreads();
    if (tid < 32) {               // attention, one thread per batch
      int bq = tid;
      float sc[10]; float mx = -3.0e38f;
      for (int t = 0; t < 10; ++t) {
        float s = 0.f;
        for (int e = 0; e < 8; ++e) {
          float en = P.attn_b[e];
          for (int j = 0; j < 8; ++j) en += hm[bq*8 + j] * P.attn_W[e*16 + j];
          for (int j = 0; j < 8; ++j) en += eo[t*256 + bq*8 + j] * P.attn_W[e*16 + 8 + j];
          s += fmaxf(en, 0.f) * P.vv[e];
        }
        sc[t] = s; mx = fmaxf(mx, s);
      }
      float sum = 0.f;
      for (int t = 0; t < 10; ++t) { sc[t] = expf(sc[t] - mx); sum += sc[t]; }
      float inv = 1.f / sum;
      for (int h = 0; h < 8; ++h) {
        float acc = 0.f;
        for (int t = 0; t < 10; ++t) acc += sc[t] * inv * eo[t*256 + bq*8 + h];
        ctx[bq*8 + h] = acc;
      }
    }
    __syncthreads();
    xc[bb*16 + hh] = inp[tid]; xc[bb*16 + 8 + hh] = ctx[tid];
    __syncthreads();
    // decoder LSTM layer 0 (in=16)
    float g0[4];
    for (int q = 0; q < 4; ++q) {
      int row = q*8 + hh;
      float acc = P.d0bih[row] + P.d0bhh[row];
      for (int d = 0; d < 16; ++d) acc += xc[bb*16 + d] * P.d0Wih[row*16 + d];
      for (int j = 0; j < 8; ++j)  acc += hid[bb*8 + j] * P.d0Whh[row*8 + j];
      g0[q] = acc;
    }
    float c0 = sigmoidf_(g0[1]) * cel[tid] + sigmoidf_(g0[0]) * tanhf(g0[2]);
    float h0 = sigmoidf_(g0[3]) * tanhf(c0);
    __syncthreads();
    th0[tid] = h0;
    __syncthreads();
    // decoder LSTM layer 1 (in=8)
    float g1[4];
    for (int q = 0; q < 4; ++q) {
      int row = q*8 + hh;
      float acc = P.d1bih[row] + P.d1bhh[row];
      for (int d = 0; d < 8; ++d) acc += th0[bb*8 + d] * P.d1Wih[row*8 + d];
      for (int j = 0; j < 8; ++j) acc += hid[256 + bb*8 + j] * P.d1Whh[row*8 + j];
      g1[q] = acc;
    }
    float c1 = sigmoidf_(g1[1]) * cel[256 + tid] + sigmoidf_(g1[0]) * tanhf(g1[2]);
    float h1 = sigmoidf_(g1[3]) * tanhf(c1);
    __syncthreads();
    hid[tid] = h0; cel[tid] = c0; hid[256 + tid] = h1; cel[256 + tid] = c1;
    __syncthreads();
    float pr = P.fc_b[hh];
    for (int j = 0; j < 8; ++j) pr += hid[256 + bb*8 + j] * P.fc_W[hh*8 + j];
    pred[tid] = pr;
    __syncthreads();
    if (tid < 64) {
      int bq = tid >> 1, o = tid & 1;
      float acc = P.fcout_b[o];
      for (int j = 0; j < 8; ++j) acc += pred[bq*8 + j] * P.fcout_W[o*8 + j];
      out[(bq*10 + step)*2 + o] = acc;  // [B,10,2] -> (-1,2)
    }
    inp[tid] = pred[tid];
    __syncthreads();
  }
}

extern "C" void kernel_launch(void* const* d_in, const int* in_sizes, int n_in,
                              void* d_out, int out_size, void* d_ws, size_t ws_size,
                              hipStream_t stream)
{
  auto F = [&](int i) { return (const float*)d_in[i]; };
  const float* X = F(0);
  float* feats = (float*)d_ws;   // [10][32][8]

  // Param index maps. JAX pytree flatten sorts dict keys; fall back to
  // insertion-order flatten if the size fingerprint says otherwise.
  int iThW, iThb, iPhW, iPhb, iG1W, iG1b, iG2W, iG2b, iLng, iLnb, iPw, iPb;
  SeqParams P;
  if (n_in > 1 && in_sizes[1] == 128) {
    // sorted keys: X, dec{attn_W,attn_b,fc_W,fc_b,l0{Whh,Wih,bhh,bih},l1{...},v},
    //              enc{l0b,l0f,l1b,l1f each {Whh,Wih,bhh,bih}}, fcout_W, fcout_b,
    //              frames{g1W,g1b,g2W,g2b,ln_b,ln_g,pW,pb,phW,phb,thW,thb}
    P.attn_W=F(1); P.attn_b=F(2); P.fc_W=F(3); P.fc_b=F(4);
    P.d0Whh=F(5);  P.d0Wih=F(6);  P.d0bhh=F(7);  P.d0bih=F(8);
    P.d1Whh=F(9);  P.d1Wih=F(10); P.d1bhh=F(11); P.d1bih=F(12);
    P.vv=F(13);
    P.e0bWhh=F(14); P.e0bWih=F(15); P.e0bbhh=F(16); P.e0bbih=F(17);
    P.e0fWhh=F(18); P.e0fWih=F(19); P.e0fbhh=F(20); P.e0fbih=F(21);
    P.e1bWhh=F(22); P.e1bWih=F(23); P.e1bbhh=F(24); P.e1bbih=F(25);
    P.e1fWhh=F(26); P.e1fWih=F(27); P.e1fbhh=F(28); P.e1fbih=F(29);
    P.fcout_W=F(30); P.fcout_b=F(31);
    iG1W=32; iG1b=33; iG2W=34; iG2b=35; iLnb=36; iLng=37;
    iPw=38; iPb=39; iPhW=40; iPhb=41; iThW=42; iThb=43;
  } else {
    // insertion order: X, frames{thW,thb,phW,phb,g1W,g1b,g2W,g2b,ln_g,ln_b,pW,pb},
    //                  enc{l0f,l0b,l1f,l1b each {Wih,Whh,bih,bhh}},
    //                  dec{attn_W,attn_b,v,l0{Wih,Whh,bih,bhh},l1{...},fc_W,fc_b},
    //                  fcout_W, fcout_b
    iThW=1; iThb=2; iPhW=3; iPhb=4; iG1W=5; iG1b=6; iG2W=7; iG2b=8;
    iLng=9; iLnb=10; iPw=11; iPb=12;
    P.e0fWih=F(13); P.e0fWhh=F(14); P.e0fbih=F(15); P.e0fbhh=F(16);
    P.e0bWih=F(17); P.e0bWhh=F(18); P.e0bbih=F(19); P.e0bbhh=F(20);
    P.e1fWih=F(21); P.e1fWhh=F(22); P.e1fbih=F(23); P.e1fbhh=F(24);
    P.e1bWih=F(25); P.e1bWhh=F(26); P.e1bbih=F(27); P.e1bbhh=F(28);
    P.attn_W=F(29); P.attn_b=F(30); P.vv=F(31);
    P.d0Wih=F(32); P.d0Whh=F(33); P.d0bih=F(34); P.d0bhh=F(35);
    P.d1Wih=F(36); P.d1Whh=F(37); P.d1bih=F(38); P.d1bhh=F(39);
    P.fc_W=F(40); P.fc_b=F(41); P.fcout_W=F(42); P.fcout_b=F(43);
  }

  // Dynamic LDS: floats 10244*4 + halves (256*32 + 32*PHS + 256*16 + 256*16 + 256*ADJS)*2
  const size_t halves = (size_t)256*32 + (size_t)32*PHS + (size_t)256*16
                      + (size_t)256*16 + (size_t)256*ADJS;
  const size_t sm1 = (size_t)10244*4 + halves*2;   // ~226 KB < 320 KB WGP LDS
  (void)hipFuncSetAttribute(reinterpret_cast<const void*>(frame_gcn_kernel),
                            hipFuncAttributeMaxDynamicSharedMemorySize, (int)sm1);
  frame_gcn_kernel<<<dim3(32, 10), 256, sm1, stream>>>(
      X, F(iThW), F(iThb), F(iPhW), F(iPhb), F(iG1W), F(iG1b), F(iG2W), F(iG2b),
      F(iLng), F(iLnb), F(iPw), F(iPb), feats);

  const size_t sm2 = (size_t)26176*4 + 256;
  (void)hipFuncSetAttribute(reinterpret_cast<const void*>(seq_kernel),
                            hipFuncAttributeMaxDynamicSharedMemorySize, (int)sm2);
  seq_kernel<<<1, 256, sm2, stream>>>(feats, P, (float*)d_out);
}
